// MultiHeadAttention_64768106823939
// MI455X (gfx1250) — matrix operand order repaired and checked by reading, not compiled, not run on hardware
//
#include <hip/hip_runtime.h>
#include <hip/hip_bf16.h>

typedef __bf16 bf16_t;
typedef __attribute__((ext_vector_type(16))) __bf16 v16bf;
typedef __attribute__((ext_vector_type(8)))  __bf16 v8bf;
typedef __attribute__((ext_vector_type(8)))  float  v8f;

#define B_  2
#define S_  2048
#define D_  1024
#define H_  16
#define DK_ 64

__device__ __forceinline__ v8f wmma_bf16(v16bf a, v16bf b, v8f c) {
  return __builtin_amdgcn_wmma_f32_16x16x32_bf16(false, a, false, b, (short)0, c,
                                                 false, false);
}

// Build a v16bf from two contiguous 8-element bf16 chunks (2 x 16B loads).
__device__ __forceinline__ v16bf pack16(const bf16_t* lo, const bf16_t* hi) {
  v8bf a = *(const v8bf*)lo;
  v8bf b = *(const v8bf*)hi;
  return __builtin_shufflevector(a, b, 0, 1, 2, 3, 4, 5, 6, 7,
                                 8, 9, 10, 11, 12, 13, 14, 15);
}

// ---------------------------------------------------------------------------
// Kernel 0: one-time fp32->bf16 conversions.
//   Wt [3072][1024] = [Wq^T ; Wk^T ; Wv^T]   (B-operand: K contiguous per row)
//   Wot[1024][1024] = Wo^T
//   Qbf[4096][1024] = Q                      (A-operand: K contiguous per row)
// ---------------------------------------------------------------------------
__global__ __launch_bounds__(256) void mha_prep(
    const float* __restrict__ Wq, const float* __restrict__ Wk,
    const float* __restrict__ Wv, const float* __restrict__ Wo,
    const float* __restrict__ Q, bf16_t* __restrict__ Wt,
    bf16_t* __restrict__ Wot, bf16_t* __restrict__ Qbf) {
  size_t idx = (size_t)blockIdx.x * 256 + threadIdx.x;
  const size_t n1 = (size_t)3072 * 1024;
  const size_t n2 = n1 + (size_t)1024 * 1024;
  if (idx < n1) {
    int n = (int)(idx >> 10), kk = (int)(idx & 1023);
    int proj = n >> 10, nn = n & 1023;
    const float* W = proj == 0 ? Wq : (proj == 1 ? Wk : Wv);
    Wt[idx] = (bf16_t)W[(size_t)kk * 1024 + nn];
  } else if (idx < n2) {
    size_t j = idx - n1;
    int n = (int)(j >> 10), kk = (int)(j & 1023);
    Wot[j] = (bf16_t)Wo[(size_t)kk * 1024 + n];
  } else {
    size_t j = idx - n2;  // [0, 4096*1024)
    Qbf[j] = (bf16_t)Q[j];
  }
}

// ---------------------------------------------------------------------------
// Kernel 1: QKV projection.  M = B*S = 4096, N = 3*1024, K = 1024.
// Grid (24, 256), block 128 (4 waves; each wave owns a 16x32 output tile:
// one A fragment feeds two WMMAs per K-step).
// q,k stored [B,H,S,DK] (K-dim contiguous); v stored transposed [B,H,DK,S].
// ---------------------------------------------------------------------------
__global__ __launch_bounds__(128) void mha_qkv_gemm(
    const bf16_t* __restrict__ Qbf, const bf16_t* __restrict__ Wt,
    const float* __restrict__ bq, const float* __restrict__ bk,
    const float* __restrict__ bv, bf16_t* __restrict__ qo,
    bf16_t* __restrict__ ko, bf16_t* __restrict__ vto) {
  const int lane = threadIdx.x & 31;
  const int wave = threadIdx.x >> 5;
  const int half = lane >> 4;
  const int l16  = lane & 15;
  const int n0 = blockIdx.x * 128 + wave * 32;  // [0,3072), 2 col-tiles
  const int m0 = blockIdx.y * 16;               // [0,4096)

  v8f acc0 = {}, acc1 = {};
  const bf16_t* arow  = Qbf + (size_t)(m0 + l16) * D_;
  const bf16_t* brow0 = Wt + (size_t)(n0 + l16) * D_;
  const bf16_t* brow1 = Wt + (size_t)(n0 + 16 + l16) * D_;
  for (int kb = 0; kb < D_; kb += 32) {
    __builtin_prefetch(brow0 + kb + 256, 0, 0);
    __builtin_prefetch(brow1 + kb + 256, 0, 0);
    const bf16_t* ap = arow + kb + half * 16;
    v16bf a  = pack16(ap, ap + 8);
    v16bf b0 = *(const v16bf*)(brow0 + kb + half * 16);
    v16bf b1 = *(const v16bf*)(brow1 + kb + half * 16);
    acc0 = wmma_bf16(a, b0, acc0);
    acc1 = wmma_bf16(a, b1, acc1);
  }

#pragma unroll
  for (int t = 0; t < 2; ++t) {
    const v8f acc = t == 0 ? acc0 : acc1;
    const int nglob = n0 + t * 16 + l16;
    const int proj = nglob >> 10;
    const int nn = nglob & 1023;
    const int h = nn >> 6, dk = nn & 63;
    const float bias = (proj == 0 ? bq : (proj == 1 ? bk : bv))[nn];
    bf16_t* dst = (proj == 0 ? qo : (proj == 1 ? ko : vto));
#pragma unroll
    for (int g = 0; g < 8; ++g) {
      int m = m0 + g + half * 8;
      int bidx = m >> 11, s = m & 2047;
      bf16_t hv = (bf16_t)(acc[g] + bias);
      size_t off;
      if (proj == 2)
        off = ((size_t)(bidx * H_ + h) * DK_ + dk) * S_ + s;  // v transposed
      else
        off = ((size_t)(bidx * H_ + h) * S_ + s) * DK_ + dk;
      dst[off] = hv;
    }
  }
}

// ---------------------------------------------------------------------------
// Kernel 2: attention.  One workgroup (8 waves / 256 threads) per
// (bh in [0,32), 16-query tile).  Full 16x2048 score block lives in LDS
// (128 KB of the 320 KB WGP LDS); attn tensor touches HBM exactly once.
// ---------------------------------------------------------------------------
__global__ __launch_bounds__(256) void mha_attention(
    const bf16_t* __restrict__ qb, const bf16_t* __restrict__ kb_,
    const bf16_t* __restrict__ vt, const unsigned char* __restrict__ mask,
    float* __restrict__ attn_out, bf16_t* __restrict__ ctx_out) {
  extern __shared__ char smem[];
  float* s_sc  = (float*)smem;          // 16 * 2048 f32 = 128 KB
  float* s_ctx = s_sc + 16 * S_;        // 16 * 64  f32 =   4 KB

  const int tid = threadIdx.x;
  const int lane = tid & 31, wave = tid >> 5;
  const int half = lane >> 4, l16 = lane & 15;
  const int qt = blockIdx.x;            // [0,128)
  const int bh = blockIdx.y;            // [0,32)
  const int b  = bh >> 4, h = bh & 15;
  const int q0 = qt * 16;
  const int key_lo = wave * 256;        // this wave's key slice

#pragma unroll
  for (int i = 0; i < 4; ++i) s_ctx[tid + 256 * i] = 0.0f;

  // ---- Phase 1: scores -> LDS (scaled) ----
  const bf16_t* qrow = qb + ((size_t)bh * S_ + q0 + l16) * DK_ + half * 16;
  v16bf aq0 = pack16(qrow,      qrow + 8);    // K = 0..31
  v16bf aq1 = pack16(qrow + 32, qrow + 40);   // K = 32..63

  for (int kt = 0; kt < 16; ++kt) {
    const int key0 = key_lo + kt * 16;
    const bf16_t* krow = kb_ + ((size_t)bh * S_ + key0 + l16) * DK_;
    v16bf b0 = *(const v16bf*)(krow + half * 16);
    v16bf b1 = *(const v16bf*)(krow + 32 + half * 16);
    v8f acc = {};
    acc = wmma_bf16(aq0, b0, acc);
    acc = wmma_bf16(aq1, b1, acc);
#pragma unroll
    for (int g = 0; g < 8; ++g)
      s_sc[(size_t)(g + half * 8) * S_ + key0 + l16] = acc[g] * 0.125f;
  }
  __syncthreads();

  // ---- Phase 2: mask + softmax; write attn to global (once) ----
#pragma unroll
  for (int rr = 0; rr < 2; ++rr) {
    const int r  = wave * 2 + rr;       // row owned exclusively by this wave
    const int qg = q0 + r;
    const unsigned char* mk = mask + ((size_t)b * S_ + qg) * S_;
    float* row = s_sc + (size_t)r * S_;

    float mx = -3.4e38f;
    for (int j = lane; j < S_; j += 32) {
      float v = mk[j] ? -1.0e9f : row[j];
      row[j] = v;
      mx = fmaxf(mx, v);
    }
#pragma unroll
    for (int off = 16; off; off >>= 1) mx = fmaxf(mx, __shfl_xor(mx, off, 32));

    float sum = 0.0f;
    for (int j = lane; j < S_; j += 32) {
      float e = __expf(row[j] - mx);
      row[j] = e;
      sum += e;
    }
#pragma unroll
    for (int off = 16; off; off >>= 1) sum += __shfl_xor(sum, off, 32);
    const float inv = 1.0f / sum;

    float* aout = attn_out + ((size_t)bh * S_ + qg) * S_;
    for (int j = lane; j < S_; j += 32) {
      float a = row[j] * inv;
      row[j] = a;
      aout[j] = a;
    }
  }
  __syncthreads();

  // ---- Phase 3: context = attn @ v (per-wave partial, ds_add_f32 reduce) ----
  const bf16_t* vbase = vt + (size_t)bh * DK_ * S_;
#pragma unroll
  for (int t = 0; t < 4; ++t) {
    const int dk0 = t * 16;
    v8f acc = {};
    for (int kc = 0; kc < 8; ++kc) {
      const int kb = key_lo + kc * 32;
      const float* ar = s_sc + (size_t)l16 * S_ + kb + half * 16;
      v16bf a;
#pragma unroll
      for (int i = 0; i < 8; ++i) {
        a[i]     = (bf16_t)ar[i];
        a[8 + i] = (bf16_t)ar[8 + i];
      }
      v16bf bm = *(const v16bf*)(vbase + (size_t)(dk0 + l16) * S_ + kb + half * 16);
      acc = wmma_bf16(a, bm, acc);
    }
#pragma unroll
    for (int g = 0; g < 8; ++g)
      atomicAdd(&s_ctx[(g + half * 8) * 64 + dk0 + l16], acc[g]);
  }
  __syncthreads();

  // ---- Phase 4: context -> bf16 workspace [B*S][H*DK] ----
#pragma unroll
  for (int i = 0; i < 4; ++i) {
    int e = tid + 256 * i;
    int r = e >> 6, c = e & 63;
    ctx_out[(size_t)(b * S_ + q0 + r) * (H_ * DK_) + h * DK_ + c] =
        (bf16_t)s_ctx[e];
  }
}

// ---------------------------------------------------------------------------
// Kernel 3: output projection.  M=4096, N=1024, K=1024.
// Grid (8, 256), block 128; each wave computes a 16x32 tile.
// ---------------------------------------------------------------------------
__global__ __launch_bounds__(128) void mha_out_gemm(
    const bf16_t* __restrict__ ctx, const bf16_t* __restrict__ Wot,
    const float* __restrict__ bo, float* __restrict__ proj) {
  const int lane = threadIdx.x & 31;
  const int wave = threadIdx.x >> 5;
  const int half = lane >> 4;
  const int l16  = lane & 15;
  const int n0 = blockIdx.x * 128 + wave * 32;
  const int m0 = blockIdx.y * 16;

  v8f acc0 = {}, acc1 = {};
  const bf16_t* arow  = ctx + (size_t)(m0 + l16) * 1024;
  const bf16_t* brow0 = Wot + (size_t)(n0 + l16) * 1024;
  const bf16_t* brow1 = Wot + (size_t)(n0 + 16 + l16) * 1024;
  for (int kb = 0; kb < 1024; kb += 32) {
    __builtin_prefetch(brow0 + kb + 256, 0, 0);
    __builtin_prefetch(brow1 + kb + 256, 0, 0);
    const bf16_t* ap = arow + kb + half * 16;
    v16bf a  = pack16(ap, ap + 8);
    v16bf b0 = *(const v16bf*)(brow0 + kb + half * 16);
    v16bf b1 = *(const v16bf*)(brow1 + kb + half * 16);
    acc0 = wmma_bf16(a, b0, acc0);
    acc1 = wmma_bf16(a, b1, acc1);
  }
#pragma unroll
  for (int t = 0; t < 2; ++t) {
    const v8f acc = t == 0 ? acc0 : acc1;
    const int n = n0 + t * 16 + l16;
    const float bias = bo[n];
#pragma unroll
    for (int g = 0; g < 8; ++g)
      proj[(size_t)(m0 + g + half * 8) * 1024 + n] = acc[g] + bias;
  }
}

// ---------------------------------------------------------------------------
// Kernel 4: residual + LayerNorm.  One row (1024) per 256-thread block.
// ---------------------------------------------------------------------------
__global__ __launch_bounds__(256) void mha_layernorm(
    const float* __restrict__ proj, const float* __restrict__ Q,
    const float* __restrict__ g, const float* __restrict__ bb,
    float* __restrict__ out) {
  const int row = blockIdx.x;  // [0,4096)
  const int lane = threadIdx.x & 31, wave = threadIdx.x >> 5;
  const float* pr = proj + (size_t)row * 1024;
  const float* qr = Q + (size_t)row * 1024;

  float x[4], s = 0.0f, s2 = 0.0f;
#pragma unroll
  for (int i = 0; i < 4; ++i) {
    int c = threadIdx.x + 256 * i;
    float v = pr[c] + qr[c];
    x[i] = v;
    s += v;
    s2 += v * v;
  }
#pragma unroll
  for (int off = 16; off; off >>= 1) {
    s  += __shfl_xor(s, off, 32);
    s2 += __shfl_xor(s2, off, 32);
  }
  __shared__ float rs[8], rs2[8];
  if (lane == 0) { rs[wave] = s; rs2[wave] = s2; }
  __syncthreads();
  if (wave == 0) {
    float a = (lane < 8) ? rs[lane] : 0.0f;
    float b2 = (lane < 8) ? rs2[lane] : 0.0f;
#pragma unroll
    for (int off = 4; off; off >>= 1) {
      a  += __shfl_xor(a, off, 32);
      b2 += __shfl_xor(b2, off, 32);
    }
    if (lane == 0) { rs[0] = a; rs2[0] = b2; }
  }
  __syncthreads();
  const float mu  = rs[0] * (1.0f / 1024.0f);
  const float var = rs2[0] * (1.0f / 1024.0f) - mu * mu;
  const float inv = rsqrtf(var + 1e-5f);
#pragma unroll
  for (int i = 0; i < 4; ++i) {
    int c = threadIdx.x + 256 * i;
    out[(size_t)row * 1024 + c] = (x[i] - mu) * inv * g[c] + bb[c];
  }
}

// ---------------------------------------------------------------------------
extern "C" void kernel_launch(void* const* d_in, const int* in_sizes, int n_in,
                              void* d_out, int out_size, void* d_ws,
                              size_t ws_size, hipStream_t stream) {
  (void)in_sizes; (void)n_in; (void)out_size; (void)ws_size;
  const float* Q  = (const float*)d_in[0];
  // d_in[1] (K) and d_in[2] (V) are unused: the reference projects from Q.
  const unsigned char* mask = (const unsigned char*)d_in[3];
  const float* Wq = (const float*)d_in[4];
  const float* bq = (const float*)d_in[5];
  const float* Wk = (const float*)d_in[6];
  const float* bk = (const float*)d_in[7];
  const float* Wv = (const float*)d_in[8];
  const float* bv = (const float*)d_in[9];
  const float* Wo = (const float*)d_in[10];
  const float* bo = (const float*)d_in[11];
  const float* ln_g = (const float*)d_in[12];
  const float* ln_b = (const float*)d_in[13];

  float* out_ln   = (float*)d_out;
  float* out_attn = out_ln + (size_t)B_ * S_ * D_;  // attn starts after LN out

  char* ws = (char*)d_ws;
  bf16_t* Wt   = (bf16_t*)ws; ws += (size_t)3072 * 1024 * sizeof(bf16_t);
  bf16_t* Wot  = (bf16_t*)ws; ws += (size_t)1024 * 1024 * sizeof(bf16_t);
  bf16_t* Qbf  = (bf16_t*)ws; ws += (size_t)B_ * S_ * D_ * sizeof(bf16_t);
  bf16_t* qb   = (bf16_t*)ws; ws += (size_t)B_ * H_ * S_ * DK_ * sizeof(bf16_t);
  bf16_t* kbuf = (bf16_t*)ws; ws += (size_t)B_ * H_ * S_ * DK_ * sizeof(bf16_t);
  bf16_t* vtb  = (bf16_t*)ws; ws += (size_t)B_ * H_ * DK_ * S_ * sizeof(bf16_t);
  bf16_t* ctxb = (bf16_t*)ws; ws += (size_t)B_ * S_ * H_ * DK_ * sizeof(bf16_t);
  float*  projb = (float*)ws;  // B*S*D f32 = 16 MB

  // 3072*1024 + 1024*1024 + 4096*1024 = 8,388,608 elems / 256 = 32768 blocks
  mha_prep<<<32768, 256, 0, stream>>>(Wq, Wk, Wv, Wo, Q, Wt, Wot, Qbf);

  mha_qkv_gemm<<<dim3(24, 256), 128, 0, stream>>>(Qbf, Wt, bq, bk, bv, qb,
                                                  kbuf, vtb);

  const size_t lds_bytes = (size_t)(16 * S_ + 16 * 64) * sizeof(float);  // 132 KB
  mha_attention<<<dim3(128, 32), 256, lds_bytes, stream>>>(qb, kbuf, vtb, mask,
                                                           out_attn, ctxb);

  mha_out_gemm<<<dim3(8, 256), 128, 0, stream>>>(ctxb, Wot, bo, projb);

  mha_layernorm<<<4096, 256, 0, stream>>>(projb, Q, ln_g, ln_b, out_ln);
}
